// Gate_16252156248243
// MI455X (gfx1250) — matrix operand order repaired and checked by reading, not compiled, not run on hardware
//
#include <hip/hip_runtime.h>
#include <stdint.h>

// ---- problem constants --------------------------------------------------
#define TOKENS 16384
#define DIM    7168
#define NEXP   256
#define TOPK   8
#define RSCALE 2.5f

// ---- tiling -------------------------------------------------------------
#define BM 64          // tokens per block
#define KC 32          // K chunk (one bf16 WMMA K-step)
#define XS 20          // uint32 stride per x-row in LDS (40 bf16, padded, 16B aligned)
#define WS 20          // uint32 stride per w-row in LDS (TDM pad reproduces this)
#define SS 256         // float stride for score rows

typedef __attribute__((ext_vector_type(16))) __bf16       v16bf;
typedef __attribute__((ext_vector_type(8)))  float        v8f;
typedef __attribute__((ext_vector_type(4)))  unsigned int u32x4;
typedef __attribute__((ext_vector_type(4)))  unsigned int v4u;
typedef __attribute__((ext_vector_type(8)))  int          v8i;
typedef __attribute__((ext_vector_type(4)))  int          v4i;

union FragU { u32x4 q[2]; v16bf v; };

struct Stage {
  unsigned int xhi[BM * XS];      //  5120 B
  unsigned int xlo[BM * XS];      //  5120 B
  unsigned int whi[NEXP * WS];    // 20480 B (TDM destination, padded rows)
  unsigned int wlo[NEXP * WS];    // 20480 B (TDM destination, padded rows)
};
union SMem {
  Stage st;
  float sc[BM * SS];              // 65536 B: sigmoid scores for fused top-k
};

__device__ __forceinline__ unsigned short f32_to_bf16_rne(float f) {
  unsigned int u = __float_as_uint(f);
  unsigned int lsb = (u >> 16) & 1u;
  u += 0x7fffu + lsb;                       // round-to-nearest-even
  return (unsigned short)(u >> 16);
}
__device__ __forceinline__ float bf16_to_f32(unsigned short h) {
  return __uint_as_float(((unsigned int)h) << 16);
}

// split one float4 into packed bf16 hi-pair and lo-pair (residual) uints
__device__ __forceinline__ void split4(float4 f, unsigned int* h, unsigned int* l) {
  float vf[4] = { f.x, f.y, f.z, f.w };
  unsigned short hh[4], ll[4];
#pragma unroll
  for (int i = 0; i < 4; ++i) {
    unsigned short hb = f32_to_bf16_rne(vf[i]);
    float rem = vf[i] - bf16_to_f32(hb);
    hh[i] = hb;
    ll[i] = f32_to_bf16_rne(rem);
  }
  h[0] = (unsigned int)hh[0] | ((unsigned int)hh[1] << 16);
  h[1] = (unsigned int)hh[2] | ((unsigned int)hh[3] << 16);
  l[0] = (unsigned int)ll[0] | ((unsigned int)ll[1] << 16);
  l[1] = (unsigned int)ll[2] | ((unsigned int)ll[3] << 16);
}

// ============ prep kernel: split W into bf16 hi/lo once (into d_ws) ======
__global__ __launch_bounds__(256)
void wsplit_kernel(const float* __restrict__ w,
                   unsigned int* __restrict__ hiPk,   // [NEXP*DIM/2] packed bf16 pairs
                   unsigned int* __restrict__ loPk) {
  const int idx = blockIdx.x * 256 + threadIdx.x;     // float4 index
  const float4 v = ((const float4*)w)[idx];
  unsigned int h[2], l[2];
  split4(v, h, l);
  hiPk[idx * 2 + 0] = h[0]; hiPk[idx * 2 + 1] = h[1];
  loPk[idx * 2 + 0] = l[0]; loPk[idx * 2 + 1] = l[1];
}

// ============ TDM: 2D tile (256 rows x 64B, stride 14336B) -> padded LDS ==
#if __has_builtin(__builtin_amdgcn_tensor_load_to_lds)
#define USE_TDM 1
__device__ __forceinline__ void tdm_load_wtile(unsigned int lds_off,
                                               const void* gaddr) {
  const unsigned long long ga = (unsigned long long)(uintptr_t)gaddr;
  v4u g0; v8i g1;
  v4i g2 = (v4i)0;
  v4i g3 = (v4i)0;
  v8i g4 = (v8i)0;                               // extra words (clang-23 6-arg form)
  g0[0] = 1u;                                    // count=1, user descriptor
  g0[1] = lds_off;                               // lds_addr (bytes)
  g0[2] = (unsigned int)ga;                      // global_addr[31:0]
  g0[3] = (unsigned int)(ga >> 32) | (2u << 30); // global_addr[56:32] | type=2
  // data_size=8B(3); pad_enable; pad_interval=16 DWORDs(3); pad_amount=4 DWORDs(3)
  g1[0] = (int)((3u << 16) | (1u << 20) | (3u << 22) | (3u << 25));
  g1[1] = (int)(8u << 16);                       // tensor_dim0 = 8  (8B units)
  g1[2] = (int)(256u << 16);                     // tensor_dim1 = 256
  g1[3] = (int)(8u << 16);                       // tile_dim0   = 8  (64B row)
  g1[4] = (int)256u;                             // tile_dim1   = 256 rows
  g1[5] = (int)1792u;                            // tensor_dim0_stride = 14336B/8
  g1[6] = 0;
  g1[7] = 0;
  __builtin_amdgcn_tensor_load_to_lds(g0, g1, g2, g3, g4, 0);
}
#endif

// ============================ main kernel ================================
__global__ __launch_bounds__(256)
void moe_router_kernel(const float* __restrict__ x,
                       const unsigned int* __restrict__ whiG,  // packed bf16 [NEXP][DIM/2]
                       const unsigned int* __restrict__ wloG,
                       const float* __restrict__ bias,
                       float* __restrict__ outW,
                       int*   __restrict__ outI) {
  __shared__ __align__(16) SMem smem;

  const int tid   = threadIdx.x;
  const int lane  = tid & 31;
  const int wave  = tid >> 5;          // 0..7
  const int waveM = wave >> 2;         // 0..1 : rows  [waveM*32, +32)
  const int waveN = wave & 3;          // 0..3 : exps  [waveN*64, +64)
  const int tokBase = blockIdx.x * BM;

  const int ldrow = tid >> 3;          // 0..31, staging row within a pass
  const int seg   = tid & 7;           // float4 segment within a K-chunk

  const unsigned int whiLds = (unsigned int)(uintptr_t)(void*)smem.st.whi;
  const unsigned int wloLds = (unsigned int)(uintptr_t)(void*)smem.st.wlo;

  v8f acc[2][4] = {};                  // 32 (M) x 64 (N) f32 accumulators

  // ================= GEMM: scores = x @ W^T via bf16x3 ===================
  for (int k0 = 0; k0 < DIM; k0 += KC) {
#if defined(USE_TDM)
    // ---- W tiles via Tensor Data Mover (wave 0 issues, EXEC-independent) ----
    if (wave == 0) {
      tdm_load_wtile(whiLds, (const void*)(whiG + (size_t)k0 / 2));
      tdm_load_wtile(wloLds, (const void*)(wloG + (size_t)k0 / 2));
    }
#else
    // ---- fallback: vector copy of pre-split bf16 W tiles ----
    {
      const int e = tid;                       // one row per thread
      const unsigned int* sh = whiG + (size_t)e * (DIM / 2) + k0 / 2;
      const unsigned int* sl = wloG + (size_t)e * (DIM / 2) + k0 / 2;
#pragma unroll
      for (int q = 0; q < 4; ++q) {
        *(u32x4*)(&smem.st.whi[e * WS + q * 4]) = *(const u32x4*)(sh + q * 4);
        *(u32x4*)(&smem.st.wlo[e * WS + q * 4]) = *(const u32x4*)(sl + q * 4);
      }
    }
#endif
    // ---- stage x tile (64 x 32 f32 -> bf16 hi/lo) ----
#pragma unroll
    for (int p = 0; p < 2; ++p) {
      const int r = p * 32 + ldrow;
      const float4 v = *(const float4*)(x + (size_t)(tokBase + r) * DIM + k0 + seg * 4);
      unsigned int h[2], l[2];
      split4(v, h, l);
      const int o = r * XS + seg * 2;
      *(uint2*)(&smem.st.xhi[o]) = make_uint2(h[0], h[1]);
      *(uint2*)(&smem.st.xlo[o]) = make_uint2(l[0], l[1]);
    }
    // ---- prefetch next x chunk (global_prefetch_b8) ----
    if (k0 + KC < DIM) {
      __builtin_prefetch(x + (size_t)(tokBase + (tid & 63)) * DIM + (k0 + KC), 0, 1);
      __builtin_prefetch(x + (size_t)(tokBase + (tid & 63)) * DIM + (k0 + KC) + 16, 0, 1);
    }
#if defined(USE_TDM)
    if (wave == 0) __builtin_amdgcn_s_wait_tensorcnt(0);
#endif
    __syncthreads();

    // ---- A fragments (16x32 bf16 per ISA layout) ----
    FragU ah[2], al[2];
#pragma unroll
    for (int m = 0; m < 2; ++m) {
      const int row = waveM * 32 + m * 16 + (lane & 15);
      const int ko  = (lane >> 4) * 4;          // K offset 0/8 elems -> 0/4 uints
      const unsigned int* ph = smem.st.xhi + row * XS + ko;
      const unsigned int* pl = smem.st.xlo + row * XS + ko;
      ah[m].q[0] = *(const u32x4*)(ph);         // K +0..7
      ah[m].q[1] = *(const u32x4*)(ph + 8);     // K +16..23
      al[m].q[0] = *(const u32x4*)(pl);
      al[m].q[1] = *(const u32x4*)(pl + 8);
    }
    // ---- B fragments + 3-term WMMA ----
#pragma unroll
    for (int n = 0; n < 4; ++n) {
      const int e  = waveN * 64 + n * 16 + (lane & 15);
      const int ko = (lane >> 4) * 4;           // K offset 0/8 elems -> 0/4 uints
      FragU bh, bl;
      const unsigned int* ph = smem.st.whi + e * WS + ko;
      const unsigned int* pl = smem.st.wlo + e * WS + ko;
      bh.q[0] = *(const u32x4*)(ph);
      bh.q[1] = *(const u32x4*)(ph + 8);
      bl.q[0] = *(const u32x4*)(pl);
      bl.q[1] = *(const u32x4*)(pl + 8);
#pragma unroll
      for (int m = 0; m < 2; ++m) {
        acc[m][n] = __builtin_amdgcn_wmma_f32_16x16x32_bf16(
            false, ah[m].v, false, bh.v, (short)0, acc[m][n], false, false);
        acc[m][n] = __builtin_amdgcn_wmma_f32_16x16x32_bf16(
            false, al[m].v, false, bh.v, (short)0, acc[m][n], false, false);
        acc[m][n] = __builtin_amdgcn_wmma_f32_16x16x32_bf16(
            false, ah[m].v, false, bl.v, (short)0, acc[m][n], false, false);
      }
    }
    __syncthreads();
  }

  // ================= sigmoid -> shared score matrix ======================
#pragma unroll
  for (int m = 0; m < 2; ++m) {
    const int rbase = waveM * 32 + m * 16 + (lane >> 4) * 8;   // C/D layout rows
#pragma unroll
    for (int n = 0; n < 4; ++n) {
      const int col = waveN * 64 + n * 16 + (lane & 15);
#pragma unroll
      for (int r = 0; r < 8; ++r) {
        const float s = 1.0f / (1.0f + __expf(-acc[m][n][r]));
        smem.sc[(rbase + r) * SS + col] = s;
      }
    }
  }
  __syncthreads();

  // ================= fused biased top-8 (one wave per token) ============
  for (int i = 0; i < 8; ++i) {
    const int tl = wave * 8 + i;               // local token 0..63
    float bv[8];
#pragma unroll
    for (int j = 0; j < 8; ++j) {
      const int e = lane + 32 * j;
      bv[j] = smem.sc[tl * SS + e] + bias[e];  // bias only affects selection
    }
    float sum = 0.0f;
    int   selE[8];
    float selW[8];
#pragma unroll
    for (int r = 0; r < 8; ++r) {
      float mv = -3.4e38f; int mj = 0;
#pragma unroll
      for (int j = 0; j < 8; ++j) {
        if (bv[j] > mv) { mv = bv[j]; mj = j; }
      }
      // monotonic key: larger score wins, ties -> smaller expert index
      const unsigned int u   = __float_as_uint(mv);
      const unsigned int key = (u & 0x80000000u) ? ~u : (u | 0x80000000u);
      const int e = lane + 32 * mj;
      unsigned long long comb =
          ((unsigned long long)key << 9) | (unsigned long long)(511 - e);
#pragma unroll
      for (int off = 16; off > 0; off >>= 1) {
        const unsigned long long o = __shfl_xor(comb, off, 32);
        if (o > comb) comb = o;
      }
      const int ew = 511 - (int)(comb & 511ull);
      const float wv = smem.sc[tl * SS + ew];  // original (unbiased) score
      selE[r] = ew; selW[r] = wv; sum += wv;
      if ((ew & 31) == lane) bv[ew >> 5] = -3.4e38f;   // mask winner
    }
    int myE = 0; float myW = 0.0f;
#pragma unroll
    for (int r = 0; r < 8; ++r) {
      if (lane == r) { myE = selE[r]; myW = selW[r]; }
    }
    if (lane < TOPK) {
      const int t = tokBase + tl;
      outW[t * TOPK + lane] = RSCALE * myW / sum;
      outI[t * TOPK + lane] = myE;
    }
  }
}

extern "C" void kernel_launch(void* const* d_in, const int* in_sizes, int n_in,
                              void* d_out, int out_size, void* d_ws, size_t ws_size,
                              hipStream_t stream) {
  (void)in_sizes; (void)n_in; (void)out_size; (void)ws_size;
  const float* x    = (const float*)d_in[0];
  // d_in[1] = input_ids (unused by the reference computation)
  const float* w    = (const float*)d_in[2];
  const float* bias = (const float*)d_in[3];
  float* outW = (float*)d_out;
  int*   outI = (int*)((float*)d_out + (size_t)TOKENS * TOPK);

  // d_ws: whi bf16[NEXP*DIM] (3.67MB) then wlo bf16[NEXP*DIM]
  unsigned int* whiG = (unsigned int*)d_ws;
  unsigned int* wloG = whiG + (size_t)NEXP * DIM / 2;

  // 1) split W once: NEXP*DIM/4 float4 -> 1792 blocks x 256 threads
  wsplit_kernel<<<dim3((NEXP * DIM / 4) / 256), dim3(256), 0, stream>>>(w, whiG, wloG);
  // 2) fused router GEMM + sigmoid + biased top-8
  moe_router_kernel<<<dim3(TOKENS / BM), dim3(256), 0, stream>>>(
      x, whiG, wloG, bias, outW, outI);
}
